// VectorQuantizer_WeightCodebookLoss2_16329465659952
// MI455X (gfx1250) — matrix operand order repaired and checked by reading, not compiled, not run on hardware
//
#include <hip/hip_runtime.h>

typedef __bf16 bf16_t;
typedef __attribute__((ext_vector_type(16))) __bf16 v16bf;
typedef __attribute__((ext_vector_type(8)))  float  v8f;

static constexpr size_t OUT_ELEMS = (size_t)16 * 512 * 64 * 64;   // 33554432
static constexpr size_t LOSS_OFF  = OUT_ELEMS;
static constexpr size_t IDX_OFF   = OUT_ELEMS + 1;

// ---------------------------------------------------------------------------
// Kernel 1: split codebook into (hi, lo) bf16 and pre-swizzle into the
// B-operand order this code uses with v_wmma_f32_16x16x32_bf16 (contiguous k, same as A):
//   B chunk kc (32 K-values), lane half h (0: K=kc*32+0..15, 1: K=kc*32+16..31),
//   code n -> 16 contiguous bf16 at [((kc*2+h)*1024 + n)*16 + e], e = K&15.
// Also computes ||e_n||^2 in f32.
// ---------------------------------------------------------------------------
__global__ __launch_bounds__(128) void vq_pack(const float* __restrict__ cb,
                                               bf16_t* __restrict__ Bh,
                                               bf16_t* __restrict__ Bl,
                                               float* __restrict__ enorm) {
    const int n = blockIdx.x;      // code index 0..1023
    const int t = threadIdx.x;     // 0..127
    float s = 0.f;
    #pragma unroll
    for (int it = 0; it < 4; ++it) {
        int c = t + (it << 7);
        float val = cb[((size_t)n << 9) + c];
        s += val * val;
        bf16_t hi = (bf16_t)val;
        bf16_t lo = (bf16_t)(val - (float)hi);
        int kc = c >> 5;
        int half = (c >> 4) & 1;
        int e = c & 15;
        size_t off = ((size_t)(((kc << 1) + half) << 10) + n) * 16 + e;
        Bh[off] = hi;
        Bl[off] = lo;
    }
    __shared__ float red[128];
    red[t] = s;
    __syncthreads();
    for (int o = 64; o > 0; o >>= 1) {
        if (t < o) red[t] += red[t + o];
        __syncthreads();
    }
    if (t == 0) enorm[n] = red[0];
}

// ---------------------------------------------------------------------------
// Kernel 2: main. One block = one stripe of 16 pixels (same b, contiguous hw).
// 8 waves; wave w owns codes [w*128, w*128+128). 3-term bf16 split GEMM
// (zh*eh + zh*el + zl*eh) with f32 accumulation via v_wmma_f32_16x16x32_bf16.
// ---------------------------------------------------------------------------
__global__ __launch_bounds__(256) void vq_main(const float* __restrict__ z,
                                               const float* __restrict__ cb,
                                               const bf16_t* __restrict__ Bh,
                                               const bf16_t* __restrict__ Bl,
                                               const float* __restrict__ enorm,
                                               float* __restrict__ dout,
                                               float* __restrict__ ssum) {
    // A-operand layout of v_wmma_f32_16x16x32_bf16, hi and lo matrices.
    __shared__ alignas(32) bf16_t s_ah[16 * 32 * 16];   // 16 KB
    __shared__ alignas(32) bf16_t s_al[16 * 32 * 16];   // 16 KB
    __shared__ float s_znp[256];
    __shared__ float s_znorm[16];
    __shared__ float s_wmin[8 * 16];
    __shared__ int   s_widx[8 * 16];
    __shared__ int   s_fidx[16];
    __shared__ float s_dmin[16];

    const int tid    = threadIdx.x;
    const int stripe = blockIdx.x;          // 0..4095
    const int b      = stripe >> 8;         // batch image
    const int hw0    = (stripe & 255) << 4; // first pixel (h*64+w) of stripe
    const int p      = tid & 15;            // pixel within stripe

    // ---- Stage: coalesced z reads, split hi/lo, write A-operand layout ----
    const float* zb = z + ((size_t)b << 21) + hw0; // b*512*4096
    float zn = 0.f;
    #pragma unroll 4
    for (int it = 0; it < 32; ++it) {
        int c = (tid >> 4) + (it << 4);                 // channel 0..511
        float val = zb[((size_t)c << 12) + p];          // 64B-coalesced per 16 lanes
        zn += val * val;
        bf16_t hi = (bf16_t)val;
        bf16_t lo = (bf16_t)(val - (float)hi);
        // Invert A VGPR layout: K32 = c&31 -> (lane half, element e)
        int kc  = c >> 5, K32 = c & 31;
        int lhW = K32 >> 4;
        int e   = K32 & 15;
        int idx = (((kc << 5) + (lhW << 4) + p) << 4) + e;
        s_ah[idx] = hi;
        s_al[idx] = lo;
    }
    s_znp[tid] = zn;
    __syncthreads();
    if (tid < 16) { // per-row ||z||^2 (only needed for the loss, not argmin)
        float s = 0.f;
        #pragma unroll
        for (int j = 0; j < 16; ++j) s += s_znp[tid + (j << 4)];
        s_znorm[tid] = s;
    }

    // ---- GEMM: dot(z_row, e_n) for this wave's 128 codes -------------------
    const int lane  = tid & 31;
    const int wv    = tid >> 5;
    const int lh    = lane >> 4;  // 0/1: which K-half this lane supplies for B
    const int l16   = lane & 15;
    const int nbase = wv << 7;    // wave's first code

    v8f acc[8];
    const v8f vzero = {0.f, 0.f, 0.f, 0.f, 0.f, 0.f, 0.f, 0.f};
    #pragma unroll
    for (int t = 0; t < 8; ++t) acc[t] = vzero;

    for (int kc = 0; kc < 16; ++kc) {
        v16bf ah = *(const v16bf*)&s_ah[(((kc << 5) + lane) << 4)];
        v16bf al = *(const v16bf*)&s_al[(((kc << 5) + lane) << 4)];
        const bf16_t* bhp = Bh + (((size_t)(((kc << 1) + lh) << 10)) << 4);
        const bf16_t* blp = Bl + (((size_t)(((kc << 1) + lh) << 10)) << 4);
        #pragma unroll
        for (int t = 0; t < 8; ++t) {
            int n = nbase + (t << 4) + l16;
            v16bf bh = *(const v16bf*)(bhp + ((size_t)n << 4)); // 32B/lane, coalesced, L2-resident
            v16bf bl = *(const v16bf*)(blp + ((size_t)n << 4));
            acc[t] = __builtin_amdgcn_wmma_f32_16x16x32_bf16(false, ah, false, bh,
                                                             (short)0, acc[t], false, false);
            acc[t] = __builtin_amdgcn_wmma_f32_16x16x32_bf16(false, ah, false, bl,
                                                             (short)0, acc[t], false, false);
            acc[t] = __builtin_amdgcn_wmma_f32_16x16x32_bf16(false, al, false, bh,
                                                             (short)0, acc[t], false, false);
        }
    }

    // ---- Argmin over this wave's 128 codes (score = ||e||^2 - 2*dot) ------
    float minv[8]; int mini[8];
    #pragma unroll
    for (int r = 0; r < 8; ++r) { minv[r] = 3.4e38f; mini[r] = 0; }
    #pragma unroll
    for (int t = 0; t < 8; ++t) {
        int n = nbase + (t << 4) + l16;
        float en = enorm[n];
        #pragma unroll
        for (int r = 0; r < 8; ++r) {
            float sv = __builtin_fmaf(-2.f, acc[t][r], en);
            if (sv < minv[r]) { minv[r] = sv; mini[r] = n; }
        }
    }
    // Cross-lane reduce within each 16-lane half (same rows, different codes).
    #pragma unroll
    for (int off = 1; off <= 8; off <<= 1) {
        #pragma unroll
        for (int r = 0; r < 8; ++r) {
            float ov = __shfl_xor(minv[r], off, 32);
            int   oi = __shfl_xor(mini[r], off, 32);
            if (ov < minv[r] || (ov == minv[r] && oi < mini[r])) { minv[r] = ov; mini[r] = oi; }
        }
    }
    if (l16 == 0) { // lane 0 holds rows 0..7, lane 16 holds rows 8..15
        #pragma unroll
        for (int r = 0; r < 8; ++r) {
            int row = r + (lh << 3);
            s_wmin[(wv << 4) + row] = minv[r];
            s_widx[(wv << 4) + row] = mini[r];
        }
    }
    __syncthreads();

    // ---- Cross-wave reduce, indices out, per-stripe loss sum --------------
    if (tid < 16) {
        float best = s_wmin[tid]; int bi = s_widx[tid];
        #pragma unroll
        for (int w = 1; w < 8; ++w) {
            float v  = s_wmin[(w << 4) + tid];
            int   i2 = s_widx[(w << 4) + tid];
            if (v < best || (v == best && i2 < bi)) { best = v; bi = i2; }
        }
        s_fidx[tid] = bi;
        s_dmin[tid] = best + s_znorm[tid];  // true min distance ||z - e||^2
        dout[IDX_OFF + (((size_t)stripe) << 4) + tid] = (float)bi;
    }
    __syncthreads();
    if (tid == 0) {
        float s = 0.f;
        #pragma unroll
        for (int r = 0; r < 16; ++r) s += s_dmin[r];
        ssum[stripe] = s;
    }

    // ---- Output gather: out[b,c,h,w] = codebook[idx[p], c] ----------------
    float* ob = dout + ((size_t)b << 21) + hw0;
    #pragma unroll 4
    for (int it = 0; it < 32; ++it) {
        int c  = (tid >> 4) + (it << 4);
        int qi = s_fidx[p];
        ob[((size_t)c << 12) + p] = cb[((size_t)qi << 9) + c]; // stores 64B-coalesced
    }
}

// ---------------------------------------------------------------------------
// Kernel 3: deterministic fixed-order loss reduction over 4096 stripe sums.
// codebook_loss = (1 + 0.25) * sum(d_min) / (B*C*H*W)
// ---------------------------------------------------------------------------
__global__ __launch_bounds__(256) void vq_loss(const float* __restrict__ ssum,
                                               float* __restrict__ dout) {
    __shared__ float red[256];
    int t = threadIdx.x;
    float s = 0.f;
    #pragma unroll
    for (int j = 0; j < 16; ++j) s += ssum[t + (j << 8)];
    red[t] = s;
    __syncthreads();
    for (int o = 128; o > 0; o >>= 1) {
        if (t < o) red[t] += red[t + o];
        __syncthreads();
    }
    if (t == 0) dout[LOSS_OFF] = 1.25f * red[0] / 33554432.0f;
}

extern "C" void kernel_launch(void* const* d_in, const int* in_sizes, int n_in,
                              void* d_out, int out_size, void* d_ws, size_t ws_size,
                              hipStream_t stream) {
    (void)in_sizes; (void)n_in; (void)out_size; (void)ws_size;
    const float* z  = (const float*)d_in[0];   // (16,512,64,64) f32
    const float* cb = (const float*)d_in[1];   // (1024,512) f32

    char* w = (char*)d_ws;
    bf16_t* Bh    = (bf16_t*)(w);                       // 1 MB  swizzled hi codebook
    bf16_t* Bl    = (bf16_t*)(w + (size_t)(1u << 20));  // 1 MB  swizzled lo codebook
    float*  enorm = (float*)(w + (size_t)(2u << 20));   // 4 KB  ||e||^2
    float*  ssum  = (float*)(w + (size_t)(2u << 20) + 4096); // 16 KB stripe sums
    float*  dout  = (float*)d_out;

    vq_pack<<<1024, 128, 0, stream>>>(cb, Bh, Bl, enorm);
    vq_main<<<4096, 256, 0, stream>>>(z, cb, Bh, Bl, enorm, dout, ssum);
    vq_loss<<<1, 256, 0, stream>>>(ssum, dout);
}
